// VectorizedMACE_35175782154733
// MI455X (gfx1250) — matrix operand order repaired and checked by reading, not compiled, not run on hardware
//
#include <hip/hip_runtime.h>
#include <hip/hip_bf16.h>
#include <math.h>

typedef __attribute__((ext_vector_type(16))) _Float16 v16h;
typedef __attribute__((ext_vector_type(8)))  _Float16 v8h;
typedef __attribute__((ext_vector_type(8)))  float    v8f;

#define N_NODES 10000
#define N_EDGES 100000
#define CCH 64
#define SS 16
#define NB 8
#define HIDD 64
#define NEL 10
#define NG 64
#define NL 2
#define RCUT 5.0f
#define AVG_NEI 16.0f

__device__ __forceinline__ float silu_f(float x) { return x / (1.0f + __expf(-x)); }

// A-fragment (16x32 f16, row M fixed per lane): K = {kb..kb+7} U {16+kb..16+kb+7}, kb = 8*(lane>=16)
__device__ __forceinline__ v16h load_a_frag(const _Float16* row_base, int kbase, int lane) {
  int sub = (lane >> 4) << 3;
  const _Float16* p = row_base + kbase + sub;
  v8h lo = *(const v8h*)(p);
  v8h hi = *(const v8h*)(p + 16);
  v16h f;
#pragma unroll
  for (int i = 0; i < 8; ++i) { f[i] = lo[i]; f[i + 8] = hi[i]; }
  return f;
}
// B-fragment (32x16 f16, col N = lane&15): K = {kb..kb+7} U {16+kb..16+kb+7}, kb = 8*(lane>=16), [col][k] storage
__device__ __forceinline__ v16h load_b_frag(const _Float16* col_base, int kbase, int lane) {
  int sub = (lane >> 4) << 3;
  const _Float16* p = col_base + kbase + sub;
  v8h lo = *(const v8h*)(p);
  v8h hi = *(const v8h*)(p + 16);
  v16h f;
#pragma unroll
  for (int i = 0; i < 8; ++i) { f[i] = lo[i]; f[i + 8] = hi[i]; }
  return f;
}

// ---------------- utility kernels ----------------
__global__ void zero_f32_kernel(float* p, int n) {
  for (int i = blockIdx.x * blockDim.x + threadIdx.x; i < n; i += gridDim.x * blockDim.x)
    p[i] = 0.0f;
}

// src[r][c] f32 -> dst[c][r] f16
__global__ void t_convert_kernel(const float* __restrict__ src, _Float16* __restrict__ dst,
                                 int rows, int cols) {
  int n = rows * cols;
  for (int i = blockIdx.x * blockDim.x + threadIdx.x; i < n; i += gridDim.x * blockDim.x) {
    int r = i / cols, c = i - r * cols;
    dst[c * rows + r] = (_Float16)src[i];
  }
}

__global__ void cvt_h_kernel(const float* __restrict__ src, _Float16* __restrict__ dst, int n) {
  for (int i = blockIdx.x * blockDim.x + threadIdx.x; i < n; i += gridDim.x * blockDim.x)
    dst[i] = (_Float16)src[i];
}

// A[n][c][s] f32 -> Aht[n][s][c] f16  (per-node 64x16 transpose)
__global__ void convert_A_kernel(const float* __restrict__ A, _Float16* __restrict__ Aht) {
  int n = N_NODES * 1024;
  for (int i = blockIdx.x * blockDim.x + threadIdx.x; i < n; i += gridDim.x * blockDim.x) {
    int node = i >> 10, r = i & 1023;
    int s = r >> 6, c = r & 63;
    Aht[i] = (_Float16)A[node * 1024 + c * 16 + s];
  }
}

// ---------------- embedding + E0 ----------------
__global__ __launch_bounds__(CCH) void embed_kernel(
    const float* __restrict__ attrs, const float* __restrict__ W_emb,
    const float* __restrict__ AE, const int* __restrict__ batch,
    const int* __restrict__ head, float* __restrict__ nf, float* __restrict__ out) {
  int n = blockIdx.x, t = threadIdx.x;
  float a[NEL];
#pragma unroll
  for (int e = 0; e < NEL; ++e) a[e] = attrs[n * NEL + e];
  float s = 0.f;
#pragma unroll
  for (int e = 0; e < NEL; ++e) s += a[e] * W_emb[e * CCH + t];
  nf[n * CCH + t] = s;
  if (t == 0) {
    int hh = head[batch[n]];
    float e0v = 0.f;
#pragma unroll
    for (int e = 0; e < NEL; ++e) e0v += a[e] * AE[hh * NEL + e];
    atomicAdd(&out[batch[n]], e0v);
  }
}

// ---------------- edge geometry: sph l<=3 + bessel radial ----------------
__global__ __launch_bounds__(256) void edge_geom_kernel(
    const float* __restrict__ pos, const float* __restrict__ shifts,
    const int* __restrict__ sender, const int* __restrict__ receiver,
    float* __restrict__ edge_attrs, float* __restrict__ edge_feats) {
  int e = blockIdx.x * blockDim.x + threadIdx.x;
  if (e >= N_EDGES) return;
  int si = sender[e], ri = receiver[e];
  float vx = pos[ri * 3 + 0] - pos[si * 3 + 0] + shifts[e * 3 + 0];
  float vy = pos[ri * 3 + 1] - pos[si * 3 + 1] + shifts[e * 3 + 1];
  float vz = pos[ri * 3 + 2] - pos[si * 3 + 2] + shifts[e * 3 + 2];
  float r2 = vx * vx + vy * vy + vz * vz;
  float r = sqrtf(r2);
  float inv = (r > 1e-12f) ? 1.0f / r : 0.0f;
  float x = vx * inv, y = vy * inv, z = vz * inv;
  const float s3 = 1.7320508075688772f, s5 = 2.23606797749979f, s15 = 3.872983346207417f;
  const float c1 = 2.0916500663351889f;  // sqrt(35/8)
  const float c2 = 10.246950765959598f;  // sqrt(105)
  const float c3 = 1.6201851746019651f;  // sqrt(21/8)
  const float c4 = 1.3228756555322954f;  // sqrt(7/4)
  const float c5 = 5.123475382979799f;   // sqrt(105/4)
  float* at = edge_attrs + e * SS;
  at[0] = 1.0f;
  at[1] = s3 * x; at[2] = s3 * y; at[3] = s3 * z;
  at[4] = s15 * x * y; at[5] = s15 * y * z;
  at[6] = 0.5f * s5 * (3.f * z * z - 1.f);
  at[7] = s15 * x * z;
  at[8] = 0.5f * s15 * (x * x - y * y);
  at[9] = c1 * y * (3.f * x * x - y * y);
  at[10] = c2 * x * y * z;
  at[11] = c3 * y * (5.f * z * z - 1.f);
  at[12] = c4 * z * (5.f * z * z - 3.f);
  at[13] = c3 * x * (5.f * z * z - 1.f);
  at[14] = c5 * z * (x * x - y * y);
  at[15] = c1 * x * (x * x - 3.f * y * y);
  float t = r / RCUT;
  float t6 = t * t * t; t6 = t6 * t6;
  float fcut = 1.0f - 28.0f * t6 + 48.0f * t6 * t - 21.0f * t6 * t * t;
  if (r >= RCUT) fcut = 0.0f;
  float pref = sqrtf(2.0f / RCUT) * inv * fcut;
  float* ef = edge_feats + e * NB;
#pragma unroll
  for (int k = 1; k <= NB; ++k)
    ef[k - 1] = pref * sinf((float)k * 3.14159265358979323846f * r / RCUT);
}

// ---------------- generic N x 64 @ (64x64)^T f16 WMMA GEMM (nrows % 16 == 0) ----------------
__global__ __launch_bounds__(256) void gemm_n64_kernel(
    const _Float16* __restrict__ Ah, const _Float16* __restrict__ wT,
    float* __restrict__ outp, int nrows) {
  int lane = threadIdx.x & 31;
  int wv = threadIdx.x >> 5;
  int n0 = (blockIdx.x * 8 + wv) * 16;
  if (n0 >= nrows) return;
  int lc = lane & 15;
  int hi8 = (lane >> 4) << 3;
  const _Float16* abase = Ah + (n0 + lc) * 64;
  v16h a0 = load_a_frag(abase, 0, lane);
  v16h a1 = load_a_frag(abase, 32, lane);
  for (int t = 0; t < 4; ++t) {
    v16h b0 = load_b_frag(wT + (t * 16 + lc) * 64, 0, lane);
    v16h b1 = load_b_frag(wT + (t * 16 + lc) * 64, 32, lane);
    v8f acc = {};
    acc = __builtin_amdgcn_wmma_f32_16x16x32_f16(false, a0, false, b0, (short)0, acc, false, false);
    acc = __builtin_amdgcn_wmma_f32_16x16x32_f16(false, a1, false, b1, (short)0, acc, false, false);
#pragma unroll
    for (int i = 0; i < 8; ++i)
      outp[(n0 + i + hi8) * 64 + t * 16 + lc] = acc[i];
  }
}

// ---------------- fused edge MLP + tensor-product + scatter (32 edges / wave) ----------------
#define EM_WAVES 4
#define EPW 32  // edges per wave (N_EDGES % EPW == 0)
__global__ __launch_bounds__(EM_WAVES * 32) void edge_message_kernel(
    const float* __restrict__ edge_feats, const float* __restrict__ edge_attrs,
    const int* __restrict__ sender, const int* __restrict__ receiver,
    const float* __restrict__ up, const float* __restrict__ w1,
    const _Float16* __restrict__ w2T, const _Float16* __restrict__ w3T,
    float* __restrict__ A) {
  __shared__ _Float16 sh_h1[EM_WAVES][EPW * 64];
  __shared__ _Float16 sh_h2[EM_WAVES][EPW * 64];
  __shared__ _Float16 sh_up[EM_WAVES][EPW * 64];
  int lane = threadIdx.x & 31;
  int wv = threadIdx.x >> 5;
  int e0 = (blockIdx.x * EM_WAVES + wv) * EPW;
  int lc = lane & 15;
  int hi8 = (lane >> 4) << 3;
  bool tile_valid = (e0 < N_EDGES);  // tiles are whole (E % 32 == 0)

  // phase 1: h1 = silu(edge_feats @ w1)  (32x8 @ 8x64, VALU)
  for (int r = 0; r < EPW; ++r) {
    float ef[NB];
    if (tile_valid) {
#pragma unroll
      for (int k = 0; k < NB; ++k) ef[k] = edge_feats[(e0 + r) * NB + k];
    } else {
#pragma unroll
      for (int k = 0; k < NB; ++k) ef[k] = 0.0f;
    }
#pragma unroll
    for (int cc = 0; cc < 2; ++cc) {
      int c = lane + cc * 32;
      float s = 0.f;
#pragma unroll
      for (int k = 0; k < NB; ++k) s += ef[k] * w1[k * HIDD + c];
      sh_h1[wv][r * 64 + c] = (_Float16)silu_f(s);
    }
  }
  // gather up[sender] tile (32 x 64) as f16 scale factors
  for (int idx = lane; idx < EPW * 64; idx += 32) {
    int r = idx >> 6, c = idx & 63;
    sh_up[wv][idx] = (_Float16)(tile_valid ? up[sender[e0 + r] * CCH + c] : 0.0f);
  }
  __syncthreads();

  // phase 2: h2 = silu(h1 @ w2)  (2 row tiles x 4 col tiles, B-frag shared across row tiles)
  v16h ah[2][2];
#pragma unroll
  for (int rt = 0; rt < 2; ++rt) {
    const _Float16* rb = &sh_h1[wv][(rt * 16 + lc) * 64];
    ah[rt][0] = load_a_frag(rb, 0, lane);
    ah[rt][1] = load_a_frag(rb, 32, lane);
  }
  for (int t = 0; t < 4; ++t) {
    v16h b0 = load_b_frag(w2T + (t * 16 + lc) * 64, 0, lane);
    v16h b1 = load_b_frag(w2T + (t * 16 + lc) * 64, 32, lane);
#pragma unroll
    for (int rt = 0; rt < 2; ++rt) {
      v8f acc = {};
      acc = __builtin_amdgcn_wmma_f32_16x16x32_f16(false, ah[rt][0], false, b0, (short)0, acc, false, false);
      acc = __builtin_amdgcn_wmma_f32_16x16x32_f16(false, ah[rt][1], false, b1, (short)0, acc, false, false);
#pragma unroll
      for (int i = 0; i < 8; ++i)
        sh_h2[wv][(rt * 16 + i + hi8) * 64 + t * 16 + lc] = (_Float16)silu_f(acc[i]);
    }
  }
  __syncthreads();

  // per-lane constants: col tile t IS channel c; lane&15 IS spherical index s
  float attr_r[16]; int rcv_r[16];
#pragma unroll
  for (int rt = 0; rt < 2; ++rt) {
#pragma unroll
    for (int i = 0; i < 8; ++i) {
      int idx = rt * 8 + i;
      int e = e0 + rt * 16 + i + hi8;
      attr_r[idx] = tile_valid ? edge_attrs[e * SS + lc] : 0.0f;
      rcv_r[idx] = tile_valid ? receiver[e] : 0;
    }
  }
  v16h a2[2][2];
#pragma unroll
  for (int rt = 0; rt < 2; ++rt) {
    const _Float16* rb = &sh_h2[wv][(rt * 16 + lc) * 64];
    a2[rt][0] = load_a_frag(rb, 0, lane);
    a2[rt][1] = load_a_frag(rb, 32, lane);
  }
  const float inv_avg = 1.0f / AVG_NEI;
  // phase 3: tp_w tiles = h2 @ w3; fused scale + atomic scatter into A (L2-resident)
  for (int t = 0; t < 64; ++t) {
    const _Float16* wcol = w3T + (t * 16 + lc) * 64;
    if (t < 60) __builtin_prefetch(w3T + ((t + 4) * 16 + lc) * 64, 0, 1);
    v16h b0 = load_b_frag(wcol, 0, lane);
    v16h b1 = load_b_frag(wcol, 32, lane);
#pragma unroll
    for (int rt = 0; rt < 2; ++rt) {
      v8f acc = {};
      acc = __builtin_amdgcn_wmma_f32_16x16x32_f16(false, a2[rt][0], false, b0, (short)0, acc, false, false);
      acc = __builtin_amdgcn_wmma_f32_16x16x32_f16(false, a2[rt][1], false, b1, (short)0, acc, false, false);
      if (tile_valid) {
#pragma unroll
        for (int i = 0; i < 8; ++i) {
          float scale = (float)sh_up[wv][(rt * 16 + i + hi8) * 64 + t] * attr_r[rt * 8 + i] * inv_avg;
          atomicAdd(&A[rcv_r[rt * 8 + i] * 1024 + t * 16 + lc], acc[i] * scale);
        }
      }
    }
  }
}

// ---------------- Amix[n][d][s] = sum_c Wmix[c][d] * A[n][c][s]  (WMMA, 1 wave/node) ----------------
__global__ __launch_bounds__(256) void amix_kernel(
    const _Float16* __restrict__ Aht, const _Float16* __restrict__ wmixT,
    float* __restrict__ Amix) {
  int lane = threadIdx.x & 31;
  int wv = threadIdx.x >> 5;
  int n = blockIdx.x * 8 + wv;
  if (n >= N_NODES) return;
  int lc = lane & 15;
  int hi8 = (lane >> 4) << 3;
  const _Float16* bbase = Aht + n * 1024 + lc * 64;  // [s][c] layout, col = s
  v16h b0 = load_b_frag(bbase, 0, lane);
  v16h b1 = load_b_frag(bbase, 32, lane);
  for (int dt = 0; dt < 4; ++dt) {
    v16h a0 = load_a_frag(wmixT + (dt * 16 + lc) * 64, 0, lane);
    v16h a1 = load_a_frag(wmixT + (dt * 16 + lc) * 64, 32, lane);
    v8f acc = {};
    acc = __builtin_amdgcn_wmma_f32_16x16x32_f16(false, a0, false, b0, (short)0, acc, false, false);
    acc = __builtin_amdgcn_wmma_f32_16x16x32_f16(false, a1, false, b1, (short)0, acc, false, false);
#pragma unroll
    for (int i = 0; i < 8; ++i)
      Amix[n * 1024 + (dt * 16 + i + hi8) * 16 + lc] = acc[i];
  }
}

// ---------------- node polynomial update + skip + readout energy ----------------
__global__ __launch_bounds__(CCH) void node_update_kernel(
    const float* __restrict__ Amix, const float* __restrict__ attrs,
    const float* __restrict__ nf, const float* __restrict__ W_skip_l,
    const float* __restrict__ Wc_l, const float* __restrict__ W_out_l,
    const float* __restrict__ W_ro_l, const int* __restrict__ batch,
    float* __restrict__ nf_next, float* __restrict__ out) {
  __shared__ float Bsh[CCH];
  __shared__ float red[CCH];
  int n = blockIdx.x, t = threadIdx.x;
  float a[NEL];
#pragma unroll
  for (int e = 0; e < NEL; ++e) a[e] = attrs[n * NEL + e];
  float w0 = 0.f, w1v = 0.f, w2v = 0.f;
#pragma unroll
  for (int e = 0; e < NEL; ++e) {
    const float* wc = Wc_l + e * 3 * CCH;
    w0 += a[e] * wc[0 * CCH + t];
    w1v += a[e] * wc[1 * CCH + t];
    w2v += a[e] * wc[2 * CCH + t];
  }
  const float* Am = Amix + n * 1024 + t * 16;
  float p1 = Am[0];
  float p2 = 0.f;
#pragma unroll
  for (int s = 0; s < SS; ++s) { float v = Am[s]; p2 += v * v; }
  float p3 = p2 * p1;
  Bsh[t] = w0 * p1 + w1v * p2 + w2v * p3;
  float sc = 0.f;
  for (int c = 0; c < CCH; ++c) {
    float nfc = nf[n * CCH + c];
    float acc = 0.f;
#pragma unroll
    for (int e = 0; e < NEL; ++e) acc += W_skip_l[(c * NEL + e) * CCH + t] * a[e];
    sc += nfc * acc;
  }
  __syncthreads();
  float o = sc;
  for (int c = 0; c < CCH; ++c) o += Bsh[c] * W_out_l[c * CCH + t];
  nf_next[n * CCH + t] = o;
  red[t] = o * W_ro_l[t];
  __syncthreads();
  for (int st = 32; st > 0; st >>= 1) {
    if (t < st) red[t] += red[t + st];
    __syncthreads();
  }
  if (t == 0) atomicAdd(&out[batch[n]], red[0]);
}

// ---------------- host ----------------
extern "C" void kernel_launch(void* const* d_in, const int* in_sizes, int n_in,
                              void* d_out, int out_size, void* d_ws, size_t ws_size,
                              hipStream_t stream) {
  (void)in_sizes; (void)n_in; (void)out_size; (void)ws_size;
  const float* positions = (const float*)d_in[0];
  const float* node_attrs = (const float*)d_in[1];
  const float* shifts = (const float*)d_in[2];
  const float* atomic_energies = (const float*)d_in[3];
  const float* W_emb = (const float*)d_in[4];
  const float* W_up = (const float*)d_in[5];
  const float* mlp_w1 = (const float*)d_in[6];
  const float* mlp_w2 = (const float*)d_in[7];
  const float* mlp_w3 = (const float*)d_in[8];
  const float* W_mix = (const float*)d_in[9];
  const float* W_skip = (const float*)d_in[10];
  const float* Wc = (const float*)d_in[11];
  const float* W_out = (const float*)d_in[12];
  const float* W_ro = (const float*)d_in[13];
  const int* edge_index = (const int*)d_in[14];
  const int* batch = (const int*)d_in[15];
  const int* head = (const int*)d_in[16];
  const int* sender = edge_index;
  const int* receiver = edge_index + N_EDGES;
  float* out = (float*)d_out;

  char* ws = (char*)d_ws;
  size_t off = 0;
  auto wsalloc = [&](size_t bytes) -> char* {
    char* p = ws + off;
    off += (bytes + 255) & ~(size_t)255;
    return p;
  };
  float* nf0 = (float*)wsalloc((size_t)N_NODES * CCH * 4);
  float* nf1 = (float*)wsalloc((size_t)N_NODES * CCH * 4);
  float* up = (float*)wsalloc((size_t)N_NODES * CCH * 4);
  _Float16* nf_h = (_Float16*)wsalloc((size_t)N_NODES * CCH * 2);
  float* A = (float*)wsalloc((size_t)N_NODES * 1024 * 4);  // also reused as Amix
  _Float16* Aht = (_Float16*)wsalloc((size_t)N_NODES * 1024 * 2);
  float* edge_attrs = (float*)wsalloc((size_t)N_EDGES * SS * 4);
  float* edge_feats = (float*)wsalloc((size_t)N_EDGES * NB * 4);
  _Float16* wupT = (_Float16*)wsalloc((size_t)NL * 64 * 64 * 2);
  _Float16* w2T = (_Float16*)wsalloc((size_t)NL * 64 * 64 * 2);
  _Float16* w3T = (_Float16*)wsalloc((size_t)NL * 64 * 1024 * 2);
  _Float16* wmixT = (_Float16*)wsalloc((size_t)NL * 64 * 64 * 2);

  zero_f32_kernel<<<1, 64, 0, stream>>>(out, NG);

  for (int l = 0; l < NL; ++l) {
    t_convert_kernel<<<16, 256, 0, stream>>>(W_up + l * 4096, wupT + l * 4096, 64, 64);
    t_convert_kernel<<<16, 256, 0, stream>>>(mlp_w2 + l * 4096, w2T + l * 4096, 64, 64);
    t_convert_kernel<<<256, 256, 0, stream>>>(mlp_w3 + l * 65536, w3T + l * 65536, 64, 1024);
    t_convert_kernel<<<16, 256, 0, stream>>>(W_mix + l * 4096, wmixT + l * 4096, 64, 64);
  }

  embed_kernel<<<N_NODES, CCH, 0, stream>>>(node_attrs, W_emb, atomic_energies, batch, head,
                                            nf0, out);
  edge_geom_kernel<<<(N_EDGES + 255) / 256, 256, 0, stream>>>(positions, shifts, sender, receiver,
                                                              edge_attrs, edge_feats);

  float* nf_cur = nf0;
  float* nf_nxt = nf1;
  const int edge_waves = (N_EDGES + EPW - 1) / EPW;                 // 3125
  const int em_blocks = (edge_waves + EM_WAVES - 1) / EM_WAVES;     // 782
  const int gemm_blocks = ((N_NODES + 15) / 16 + 7) / 8;            // 79
  const int amix_blocks = (N_NODES + 7) / 8;                        // 1250

  for (int l = 0; l < NL; ++l) {
    cvt_h_kernel<<<1024, 256, 0, stream>>>(nf_cur, nf_h, N_NODES * CCH);
    gemm_n64_kernel<<<gemm_blocks, 256, 0, stream>>>(nf_h, wupT + l * 4096, up, N_NODES);
    zero_f32_kernel<<<8192, 256, 0, stream>>>(A, N_NODES * 1024);
    edge_message_kernel<<<em_blocks, EM_WAVES * 32, 0, stream>>>(
        edge_feats, edge_attrs, sender, receiver, up,
        mlp_w1 + l * NB * HIDD, w2T + l * 4096, w3T + l * 65536, A);
    convert_A_kernel<<<8192, 256, 0, stream>>>(A, Aht);
    amix_kernel<<<amix_blocks, 256, 0, stream>>>(Aht, wmixT + l * 4096, A /* Amix reuse */);
    node_update_kernel<<<N_NODES, CCH, 0, stream>>>(
        A, node_attrs, nf_cur, W_skip + (size_t)l * CCH * NEL * CCH,
        Wc + (size_t)l * NEL * 3 * CCH, W_out + l * 4096, W_ro + l * CCH, batch, nf_nxt, out);
    float* tmp = nf_cur; nf_cur = nf_nxt; nf_nxt = tmp;
  }
}
